// QThreeLayerCurves_74199855006026
// MI455X (gfx1250) — 8-bit matrix operand order repaired and checked by reading, compiled, not run on hardware
//
#include <hip/hip_runtime.h>
#include <math.h>

typedef int v8i __attribute__((ext_vector_type(8)));

#define NB 512
#define NT 256

struct WS_t {
  float pmax[4][NB];
  float s[4];
  float inv_s0;
  float pad[3];
  float w_s[4][64];
  float dq[4][64];
  float r[3][64];
  int   bint[4][64];
  alignas(32) int Bpk1[4][32][8];
  alignas(32) int Bpk2[2][32][8];
  alignas(32) int Bpk3[2][32][8];
  alignas(32) int Bpk4[1][32][8];
};


__device__ __forceinline__ v8i wmma_iu8(v8i a, v8i b, v8i c) {
  return __builtin_amdgcn_wmma_i32_16x16x64_iu8(true, a, true, b, c, false, false);
}

__device__ __forceinline__ int pack4b(int b0, int b1, int b2, int b3) {
  return (b0 & 255) | ((b1 & 255) << 8) | ((b2 & 255) << 16) | ((b3 & 255) << 24);
}

__device__ __forceinline__ v8i bcast8(int v) {
  v8i c;
#pragma unroll
  for (int k = 0; k < 8; k++) c[k] = v;
  return c;
}

__device__ __forceinline__ float relumax8(v8i d) {
  int m = d[0];
#pragma unroll
  for (int k = 1; k < 8; k++) m = (d[k] > m) ? d[k] : m;
  return (float)(m > 0 ? m : 0);
}

__device__ __forceinline__ int2 requant8(v8i d, float r) {
  int q[8];
#pragma unroll
  for (int k = 0; k < 8; k++) {
    int v = d[k] > 0 ? d[k] : 0;
    float f = rintf((float)v * r);
    f = fminf(f, 31.f);
    q[k] = (int)f;
  }
  return make_int2(pack4b(q[0], q[1], q[2], q[3]), pack4b(q[4], q[5], q[6], q[7]));
}

__device__ __forceinline__ void lds_store_q(char* base, int col, int hi, int2 v) {
  *(int2*)(base + col * 16 + hi * 8) = v;
}

__device__ __forceinline__ void wait_lds() {
  asm volatile("s_wait_dscnt 0" ::: "memory");
}

template <int KB>
__device__ __forceinline__ v8i lds_loadA(const char* base, int lane) {
  int m = lane & 15, hi = lane >> 4;
  const unsigned char* cb = (const unsigned char*)base;
  v8i a = {0, 0, 0, 0, 0, 0, 0, 0};
#pragma unroll
  for (int p = 0; p < KB; p++) {
    int kb = p * 16 + hi * 8;
    a[2 * p] = pack4b(cb[(kb + 0) * 16 + m], cb[(kb + 1) * 16 + m],
                      cb[(kb + 2) * 16 + m], cb[(kb + 3) * 16 + m]);
    a[2 * p + 1] = pack4b(cb[(kb + 4) * 16 + m], cb[(kb + 5) * 16 + m],
                          cb[(kb + 6) * 16 + m], cb[(kb + 7) * 16 + m]);
  }
  return a;
}

__device__ __forceinline__ float block_max(float v, float* red) {
  int tid = threadIdx.x;
  red[tid] = v;
  __syncthreads();
#pragma unroll
  for (int s = NT / 2; s > 0; s >>= 1) {
    if (tid < s) red[tid] = fmaxf(red[tid], red[tid + s]);
    __syncthreads();
  }
  return red[0];
}

__device__ __forceinline__ int qbias(float f) {
  float v = rintf(f);
  v = fminf(fmaxf(v, -32.f), 31.f);
  return (int)v;
}


__device__ void prep_layer(const float* W, int N, int K, float* wsout, int* Bpk,
                           int ntiles, signed char* sw) {
  int tid = threadIdx.x;
  if (tid < N) {
    float m = 0.f;
    for (int k = 0; k < K; k++) m = fmaxf(m, fabsf(W[tid * K + k]));
    wsout[tid] = fmaxf(m / 31.f, 1e-8f);
  }
  __syncthreads();
  for (int e = tid; e < N * K; e += blockDim.x) {
    int o = e / K;
    float v = rintf(W[e] / wsout[o]);
    v = fminf(fmaxf(v, -32.f), 31.f);
    sw[e] = (signed char)(int)v;
  }
  __syncthreads();
  int nd = ntiles * 256;
  for (int d = tid; d < nd; d += blockDim.x) {
    int t = d >> 8, rem = d & 255, ln = rem >> 3, j = rem & 7;
    int c = ln & 15, h = ln >> 4, n = t * 16 + c;
    int ks = (j >> 1) * 16 + h * 8 + (j & 1) * 4;
    int val = 0;
#pragma unroll
    for (int b = 0; b < 4; b++) {
      int k = ks + b;
      int q = 0;
      if (n < N && k < K) q = sw[n * K + k] & 255;
      val |= q << (8 * b);
    }
    Bpk[d] = val;
  }
  __syncthreads();
}

__global__ __launch_bounds__(NT) void prep_kernel(const float* W1, const float* W2,
                                                  const float* W3, const float* W4,
                                                  WS_t* ws) {
  __shared__ signed char sw[4096];
  prep_layer(W1, 64, 16, ws->w_s[0], &ws->Bpk1[0][0][0], 4, sw);
  prep_layer(W2, 32, 64, ws->w_s[1], &ws->Bpk2[0][0][0], 2, sw);
  prep_layer(W3, 32, 32, ws->w_s[2], &ws->Bpk3[0][0][0], 2, sw);
  prep_layer(W4, 5, 32, ws->w_s[3], &ws->Bpk4[0][0][0], 1, sw);
}


__global__ __launch_bounds__(NT) void absmax_kernel(const float4* __restrict__ x4,
                                                    long long n4, WS_t* ws) {
  __shared__ float red[NT];
  float m = 0.f;
  for (long long i = (long long)blockIdx.x * NT + threadIdx.x; i < n4;
       i += (long long)gridDim.x * NT) {
    float4 v = x4[i];
    m = fmaxf(m, fmaxf(fmaxf(fabsf(v.x), fabsf(v.y)), fmaxf(fabsf(v.z), fabsf(v.w))));
  }
  float b = block_max(m, red);
  if (threadIdx.x == 0) ws->pmax[0][blockIdx.x] = b;
}


template <int S>
__global__ __launch_bounds__(NT) void scale_kernel(const float* __restrict__ bias, WS_t* ws) {
  __shared__ float red[NT];
  int tid = threadIdx.x;
  float v = fmaxf(ws->pmax[S][tid], ws->pmax[S][tid + NT]);
  float m = block_max(v, red);
  float s = fmaxf(m / 31.f, 1e-8f);
  if (tid == 0) {
    ws->s[S] = s;
    if (S == 0) ws->inv_s0 = 1.f / s;
  }
  if (S == 0) {
    if (tid < 64) {
      float d = ws->w_s[0][tid] * s;
      ws->dq[0][tid] = d;
      ws->bint[0][tid] = qbias(bias[tid] / d);
    }
  } else if (S == 1) {
    if (tid < 64) ws->r[0][tid] = ws->dq[0][tid] / s;
    if (tid < 32) {
      float d = ws->w_s[1][tid] * s;
      ws->dq[1][tid] = d;
      ws->bint[1][tid] = qbias(bias[tid] / d);
    }
  } else if (S == 2) {
    if (tid < 32) {
      ws->r[1][tid] = ws->dq[1][tid] / s;
      float d = ws->w_s[2][tid] * s;
      ws->dq[2][tid] = d;
      ws->bint[2][tid] = qbias(bias[tid] / d);
    }
  } else {
    if (tid < 32) ws->r[2][tid] = ws->dq[2][tid] / s;
    if (tid < 5) {
      float d = ws->w_s[3][tid] * s;
      ws->dq[3][tid] = d;
      ws->bint[3][tid] = qbias(bias[tid] / d);
    }
  }
}


template <int STAGE>
__global__ __launch_bounds__(NT) void pass_kernel(const float* __restrict__ x,
                                                  int2* __restrict__ q0,
                                                  WS_t* __restrict__ ws,
                                                  float* __restrict__ out,
                                                  long long tiles, int useQ0) {
  __shared__ __align__(16) char smem[8 * 1024];
  __shared__ float red[NT];
  const int tid = threadIdx.x, lane = tid & 31, wv = tid >> 5;
  const int colc = lane & 15, hi = lane >> 4;
  char* myb = smem + wv * 1024;

  const long long gw = (long long)blockIdx.x * (NT / 32) + wv;
  const long long nw = (long long)gridDim.x * (NT / 32);

  const float inv_s0 = ws->inv_s0;

  v8i B1[4], C1[4];
  float dq1v[4], r1v[4];
#pragma unroll
  for (int t = 0; t < 4; t++) {
    B1[t] = *(const v8i*)ws->Bpk1[t][lane];
    C1[t] = bcast8(ws->bint[0][t * 16 + colc]);
    dq1v[t] = ws->dq[0][t * 16 + colc];
    r1v[t] = (STAGE >= 2) ? ws->r[0][t * 16 + colc] : 0.f;
  }
  v8i B2[2] = {}, C2[2] = {};
  float dq2v[2] = {0, 0}, r2v[2] = {0, 0};
  v8i B3[2] = {}, C3[2] = {};
  float dq3v[2] = {0, 0}, r3v[2] = {0, 0};
  v8i B4 = {}, C4 = {};
  float dq4v = 0.f;
  if (STAGE >= 2) {
#pragma unroll
    for (int u = 0; u < 2; u++) {
      B2[u] = *(const v8i*)ws->Bpk2[u][lane];
      C2[u] = bcast8(ws->bint[1][u * 16 + colc]);
      dq2v[u] = ws->dq[1][u * 16 + colc];
      r2v[u] = (STAGE >= 3) ? ws->r[1][u * 16 + colc] : 0.f;
    }
  }
  if (STAGE >= 3) {
#pragma unroll
    for (int u = 0; u < 2; u++) {
      B3[u] = *(const v8i*)ws->Bpk3[u][lane];
      C3[u] = bcast8(ws->bint[2][u * 16 + colc]);
      dq3v[u] = ws->dq[2][u * 16 + colc];
      r3v[u] = (STAGE >= 4) ? ws->r[2][u * 16 + colc] : 0.f;
    }
  }
  if (STAGE >= 4) {
    B4 = *(const v8i*)ws->Bpk4[0][lane];
    C4 = bcast8((colc < 5) ? ws->bint[3][colc] : 0);
    dq4v = (colc < 5) ? ws->dq[3][colc] : 0.f;
  }

  float runmax = 0.f;

  for (long long t = gw; t < tiles; t += nw) {
    int a0, a1;
    if (STAGE == 1 || !useQ0) {
      const float4* xp = (const float4*)(x + ((t * 16 + colc) * 16 + hi * 8));
      float4 f0 = xp[0], f1 = xp[1];
      float qf[8] = {f0.x, f0.y, f0.z, f0.w, f1.x, f1.y, f1.z, f1.w};
      int qi[8];
#pragma unroll
      for (int i = 0; i < 8; i++) {
        float v = rintf(qf[i] * inv_s0);
        v = fminf(fmaxf(v, -32.f), 31.f);
        qi[i] = (int)v;
      }
      a0 = pack4b(qi[0], qi[1], qi[2], qi[3]);
      a1 = pack4b(qi[4], qi[5], qi[6], qi[7]);
      if (STAGE == 1 && useQ0) q0[t * 32 + lane] = make_int2(a0, a1);
    } else {
      int2 v = q0[t * 32 + lane];
      a0 = v.x;
      a1 = v.y;
    }
    v8i A1 = {a0, a1, 0, 0, 0, 0, 0, 0};

    v8i d1[4];
#pragma unroll
    for (int u = 0; u < 4; u++) d1[u] = wmma_iu8(A1, B1[u], C1[u]);

    if (STAGE == 1) {
#pragma unroll
      for (int u = 0; u < 4; u++) runmax = fmaxf(runmax, relumax8(d1[u]) * dq1v[u]);
      continue;
    }

#pragma unroll
    for (int u = 0; u < 4; u++) lds_store_q(myb, u * 16 + colc, hi, requant8(d1[u], r1v[u]));
    wait_lds();
    v8i A2 = lds_loadA<4>(myb, lane);

    v8i d2[2];
#pragma unroll
    for (int u = 0; u < 2; u++) d2[u] = wmma_iu8(A2, B2[u], C2[u]);

    if (STAGE == 2) {
#pragma unroll
      for (int u = 0; u < 2; u++) runmax = fmaxf(runmax, relumax8(d2[u]) * dq2v[u]);
      continue;
    }

#pragma unroll
    for (int u = 0; u < 2; u++) lds_store_q(myb, u * 16 + colc, hi, requant8(d2[u], r2v[u]));
    wait_lds();
    v8i A3 = lds_loadA<2>(myb, lane);

    v8i d3[2];
#pragma unroll
    for (int u = 0; u < 2; u++) d3[u] = wmma_iu8(A3, B3[u], C3[u]);

    if (STAGE == 3) {
#pragma unroll
      for (int u = 0; u < 2; u++) runmax = fmaxf(runmax, relumax8(d3[u]) * dq3v[u]);
      continue;
    }

#pragma unroll
    for (int u = 0; u < 2; u++) lds_store_q(myb, u * 16 + colc, hi, requant8(d3[u], r3v[u]));
    wait_lds();
    v8i A4 = lds_loadA<2>(myb, lane);
    v8i d4 = wmma_iu8(A4, B4, C4);

    float* mf = (float*)myb;
    if (colc < 5) {
#pragma unroll
      for (int k = 0; k < 8; k++) mf[colc * 16 + hi * 8 + k] = (float)d4[k] * dq4v;
    }
    wait_lds();
    if (lane < 16) {
      float v0 = mf[0 * 16 + lane], v1 = mf[1 * 16 + lane], v2 = mf[2 * 16 + lane];
      float v3 = mf[3 * 16 + lane], v4 = mf[4 * 16 + lane];
      float mx = fmaxf(fmaxf(fmaxf(v0, v1), fmaxf(v2, v3)), v4);
      float e0 = __expf(v0 - mx), e1 = __expf(v1 - mx), e2 = __expf(v2 - mx);
      float e3 = __expf(v3 - mx), e4 = __expf(v4 - mx);
      float inv = 1.f / (e0 + e1 + e2 + e3 + e4);
      float* op = out + (t * 16 + lane) * 5;
      op[0] = e0 * inv;
      op[1] = e1 * inv;
      op[2] = e2 * inv;
      op[3] = e3 * inv;
      op[4] = e4 * inv;
    }
  }

  if (STAGE < 4) {
    float m = block_max(runmax, red);
    if (tid == 0) ws->pmax[STAGE][blockIdx.x] = m;
  }
}


extern "C" void kernel_launch(void* const* d_in, const int* in_sizes, int n_in,
                              void* d_out, int out_size, void* d_ws, size_t ws_size,
                              hipStream_t stream) {
  const float* x = (const float*)d_in[0];
  const float* W1 = (const float*)d_in[1];
  const float* b1 = (const float*)d_in[2];
  const float* W2 = (const float*)d_in[3];
  const float* b2 = (const float*)d_in[4];
  const float* W3 = (const float*)d_in[5];
  const float* b3 = (const float*)d_in[6];
  const float* W4 = (const float*)d_in[7];
  const float* b4 = (const float*)d_in[8];
  float* out = (float*)d_out;

  long long rows = (long long)in_sizes[0] / 16;
  long long tiles = rows / 16;
  WS_t* ws = (WS_t*)d_ws;
  size_t q0_off = (sizeof(WS_t) + 255) & ~(size_t)255;
  int useQ0 = (ws_size >= q0_off + (size_t)rows * 16) ? 1 : 0;
  int2* q0 = (int2*)((char*)d_ws + q0_off);

  prep_kernel<<<1, NT, 0, stream>>>(W1, W2, W3, W4, ws);
  absmax_kernel<<<NB, NT, 0, stream>>>((const float4*)x, rows * 4, ws);
  scale_kernel<0><<<1, NT, 0, stream>>>(b1, ws);
  pass_kernel<1><<<NB, NT, 0, stream>>>(x, q0, ws, out, tiles, useQ0);
  scale_kernel<1><<<1, NT, 0, stream>>>(b2, ws);
  pass_kernel<2><<<NB, NT, 0, stream>>>(x, q0, ws, out, tiles, useQ0);
  scale_kernel<2><<<1, NT, 0, stream>>>(b3, ws);
  pass_kernel<3><<<NB, NT, 0, stream>>>(x, q0, ws, out, tiles, useQ0);
  scale_kernel<3><<<1, NT, 0, stream>>>(b4, ws);
  pass_kernel<4><<<NB, NT, 0, stream>>>(x, q0, ws, out, tiles, useQ0);
}
